// SageTransformer_63771674411608
// MI455X (gfx1250) — matrix operand order repaired and checked by reading, not compiled, not run on hardware
//
#include <hip/hip_runtime.h>
#include <hip/hip_bf16.h>

// ---------------------------------------------------------------------------
// MI455X / gfx1250: all GEMM-shaped work via v_wmma_f32_16x16x32_bf16.
// Wave32, 128-thread blocks (4 waves). GEMM block tile 64x128 (32x64/wave).
// Causal-conv shifts handled with a zero-padded activation layout so the
// WMMA inner loops contain no divergent zero-fill.
// ---------------------------------------------------------------------------

typedef __bf16 bf16;
typedef __attribute__((ext_vector_type(8)))  __bf16 bf16x8;
typedef __attribute__((ext_vector_type(16))) __bf16 bf16x16;
typedef __attribute__((ext_vector_type(8)))  float  f32x8;

#define DEV __device__ __forceinline__

constexpr int LAYERS = 4;
constexpr int BS  = 8;
constexpr int SEQ = 512;
constexpr int DM  = 512;
constexpr int NH  = 8;
constexpr int DH  = 64;
constexpr int FF  = 2048;
constexpr int MS  = BS * SEQ;     // 4096 rows
constexpr int PAD = 8;            // zero rows in front of each batch (>= max shift 4)
constexpr int SP  = SEQ + PAD;    // padded rows per batch

DEV f32x8 zero_acc() {
  f32x8 z = {0.f, 0.f, 0.f, 0.f, 0.f, 0.f, 0.f, 0.f};
  return z;
}

// A-fragment (16x32, row per lane): half 0 holds K=0..15 contiguously,
// half 1 holds K=16..31. Caller pre-adds (lane>>4)*16.
DEV bf16x16 load_frag_a(const bf16* p) {
  bf16x8 lo = *(const bf16x8*)p;
  bf16x8 hi = *(const bf16x8*)(p + 8);
  bf16x16 r;
#pragma unroll
  for (int j = 0; j < 8; ++j) { r[j] = lo[j]; r[8 + j] = hi[j]; }
  return r;
}

// B-fragment (32x16, col per lane): half 0 holds K=0..15 contiguously,
// half 1 holds K=16..31. Caller pre-adds (lane>>4)*16.
DEV bf16x16 load_frag_b(const bf16* p) {
  bf16x8 lo = *(const bf16x8*)p;
  bf16x8 hi = *(const bf16x8*)(p + 8);
  bf16x16 r;
#pragma unroll
  for (int j = 0; j < 8; ++j) { r[j] = lo[j]; r[8 + j] = hi[j]; }
  return r;
}

DEV f32x8 wmma_bf16(bf16x16 a, bf16x16 b, f32x8 c) {
  return __builtin_amdgcn_wmma_f32_16x16x32_bf16(
      /*neg_a=*/false, a, /*neg_b=*/false, b,
      /*c_mod=*/(short)0, c, /*reuse_a=*/false, /*reuse_b=*/false);
}

// ---------------------------------------------------------------------------
// Weight transpose + fp32->bf16: Wt[n*K + k] = W[k*N + n]
// ---------------------------------------------------------------------------
__global__ void k_transpose_bf16(const float* __restrict__ W,
                                 bf16* __restrict__ Wt, int K, int N) {
  size_t idx = (size_t)blockIdx.x * 256 + threadIdx.x;
  if (idx >= (size_t)K * N) return;
  int k = (int)(idx % K);
  int n = (int)(idx / K);
  Wt[idx] = (__bf16)W[(size_t)k * N + n];
}

__global__ void k_f32_to_bf16(const float* __restrict__ in,
                              bf16* __restrict__ out, size_t n) {
  size_t i = (size_t)blockIdx.x * 256 + threadIdx.x;
  if (i < n) out[i] = (__bf16)in[i];
}

// Pack x (fp32 [B,S,D]) into zero-padded bf16 [B, PAD+S, D].
__global__ void k_pack_x_padded(const float* __restrict__ x,
                                bf16* __restrict__ xp) {
  size_t idx = (size_t)blockIdx.x * 256 + threadIdx.x;
  if (idx >= (size_t)BS * SP * DM) return;
  const int    c    = (int)(idx % DM);
  const size_t prow = idx / DM;
  const int    b    = (int)(prow / SP);
  const int    r    = (int)(prow % SP);
  const float  v = (r < PAD) ? 0.f
                             : x[((size_t)b * SEQ + (r - PAD)) * DM + c];
  xp[idx] = (__bf16)v;
}

// ---------------------------------------------------------------------------
// Causal conv as sum of row-shifted WMMA GEMMs over the padded activations.
// Xp: [BS, SP, DM] bf16 (pad rows zero). Wt: [ntaps][DM(out)][DM(in)] bf16.
// outRM: [MS, DM] bf16, or null. outVt: [BS, NH, DH, SEQ] bf16, or null.
// Block tile 64(M) x 128(N); wave tile 32 x 64. No divergence in the k-loop.
// ---------------------------------------------------------------------------
__global__ __launch_bounds__(128)
void k_conv_gemm(const bf16* __restrict__ Xp, const bf16* __restrict__ Wt,
                 const float* __restrict__ bias,
                 bf16* __restrict__ outRM, bf16* __restrict__ outVt,
                 int ntaps) {
  const int lane  = threadIdx.x & 31;
  const int wv    = threadIdx.x >> 5;
  const int wm    = wv >> 1, wn = wv & 1;
  const int mbase = blockIdx.y * 64 + wm * 32;
  const int nbase = blockIdx.x * 128 + wn * 64;
  const int l15   = lane & 15;
  const int half  = lane >> 4;
  const int aOff  = half << 4;
  const int bOff  = half << 4;

  const int bb = mbase >> 9;            // batch (tiles never straddle batches)
  const int sb = mbase & (SEQ - 1);     // seq position of first row

  f32x8 acc[2][4];
#pragma unroll
  for (int i = 0; i < 2; ++i)
#pragma unroll
    for (int j = 0; j < 4; ++j) acc[i][j] = zero_acc();

  for (int t = 0; t < ntaps; ++t) {
    const int delta = (ntaps - 1) - t;
    const bf16* Wtap = Wt + (size_t)t * DM * DM;
    // padded row index: always valid, pad rows are zeros
    const bf16* a0base =
        Xp + (size_t)(bb * SP + PAD + sb + l15 - delta) * DM + aOff;
    const bf16* a1base = a0base + (size_t)16 * DM;
#pragma unroll 2
    for (int kb = 0; kb < DM; kb += 32) {
      bf16x16 a0 = load_frag_a(a0base + kb);
      bf16x16 a1 = load_frag_a(a1base + kb);
#pragma unroll
      for (int j = 0; j < 4; ++j) {
        bf16x16 bj = load_frag_b(Wtap + (size_t)(nbase + j * 16 + l15) * DM +
                                 bOff + kb);
        acc[0][j] = wmma_bf16(a0, bj, acc[0][j]);
        acc[1][j] = wmma_bf16(a1, bj, acc[1][j]);
      }
    }
  }

#pragma unroll
  for (int i = 0; i < 2; ++i)
#pragma unroll
    for (int j = 0; j < 4; ++j)
#pragma unroll
      for (int r = 0; r < 8; ++r) {
        const int m = mbase + i * 16 + (half << 3) + r;   // C layout: row
        const int n = nbase + j * 16 + l15;               // C layout: col
        const float val = acc[i][j][r] + bias[n];
        if (outRM) outRM[(size_t)m * DM + n] = (__bf16)val;
        if (outVt) {
          const int s  = m & (SEQ - 1);
          const int b  = m >> 9;
          const int hh = n >> 6;
          const int dd = n & (DH - 1);
          outVt[(((size_t)(b * NH + hh) * DH) + dd) * SEQ + s] = (__bf16)val;
        }
      }
}

// ---------------------------------------------------------------------------
// Flash attention, one (b,h) head per blockIdx.y, one 64-row q tile per
// blockIdx.x, one branch per launch. 4 waves * 16 q-rows each.
// Q,K: [MS, DM] bf16 (head columns h*64..). Vt: [BS,NH,DH,SEQ] bf16.
// Out: fp32 [MS, DM]; first=1 overwrites (branch 0), else accumulates.
// ---------------------------------------------------------------------------
__global__ __launch_bounds__(128)
void k_attn_flash(const bf16* __restrict__ Q, const bf16* __restrict__ Kmat,
                  const bf16* __restrict__ Vt, float* __restrict__ Out,
                  int first) {
  __shared__ __bf16 ldsP[4][16 * 64];

  const int lane = threadIdx.x & 31;
  const int wv   = threadIdx.x >> 5;
  const int bh   = blockIdx.y;
  const int b    = bh >> 3;
  const int h    = bh & 7;
  const int qt   = blockIdx.x;
  const int l15  = lane & 15;
  const int half = lane >> 4;
  const int aOff = half << 4;
  const int bOff = half << 4;

  const int qrow0 = qt * 64 + wv * 16;

  bf16x16 aq[2];
  {
    const bf16* qp = Q + ((size_t)(b * SEQ + qrow0 + l15)) * DM + h * DH + aOff;
    aq[0] = load_frag_a(qp);
    aq[1] = load_frag_a(qp + 32);
  }

  f32x8 o[4];
#pragma unroll
  for (int j = 0; j < 4; ++j) o[j] = zero_acc();
  float mrow[8], lrow[8];
#pragma unroll
  for (int r = 0; r < 8; ++r) { mrow[r] = -3.0e38f; lrow[r] = 0.f; }

  const float scale = 0.125f;   // 1/sqrt(64)
  __bf16* myP = &ldsP[wv][0];

  for (int kt = 0; kt <= qt; ++kt) {
    f32x8 s4[4];
#pragma unroll
    for (int j = 0; j < 4; ++j) s4[j] = zero_acc();
#pragma unroll
    for (int kk = 0; kk < 2; ++kk) {
      const bf16* kp = Kmat + ((size_t)(b * SEQ + kt * 64 + l15)) * DM +
                       h * DH + kk * 32 + bOff;
#pragma unroll
      for (int j = 0; j < 4; ++j) {
        bf16x16 bk = load_frag_b(kp + (size_t)j * 16 * DM);
        s4[j] = wmma_bf16(aq[kk], bk, s4[j]);
      }
    }
#pragma unroll
    for (int j = 0; j < 4; ++j)
#pragma unroll
      for (int r = 0; r < 8; ++r) {
        float v = s4[j][r] * scale;
        if (kt == qt) {
          const int kg = kt * 64 + j * 16 + l15;
          const int qg = qrow0 + (half << 3) + r;
          if (kg > qg) v = -1.0e30f;
        }
        s4[j][r] = v;
      }
    float ef[8];
#pragma unroll
    for (int r = 0; r < 8; ++r) {
      float mx = fmaxf(fmaxf(s4[0][r], s4[1][r]), fmaxf(s4[2][r], s4[3][r]));
#pragma unroll
      for (int msk = 8; msk >= 1; msk >>= 1) mx = fmaxf(mx, __shfl_xor(mx, msk, 32));
      const float mn = fmaxf(mrow[r], mx);
      ef[r] = __expf(mrow[r] - mn);
      mrow[r] = mn;
    }
#pragma unroll
    for (int r = 0; r < 8; ++r) {
      float ps = 0.f;
#pragma unroll
      for (int j = 0; j < 4; ++j) {
        const float p = __expf(s4[j][r] - mrow[r]);
        ps += p;
        myP[((half << 3) + r) * 64 + j * 16 + l15] = (__bf16)p;
      }
#pragma unroll
      for (int msk = 8; msk >= 1; msk >>= 1) ps += __shfl_xor(ps, msk, 32);
      lrow[r] = lrow[r] * ef[r] + ps;
#pragma unroll
      for (int j = 0; j < 4; ++j) o[j][r] *= ef[r];
    }
#pragma unroll
    for (int kk = 0; kk < 2; ++kk) {
      bf16x16 ap = load_frag_a(myP + l15 * 64 + kk * 32 + aOff);
      const bf16* vp = Vt + (((size_t)(b * NH + h) * DH) + l15) * SEQ +
                       kt * 64 + kk * 32 + bOff;
#pragma unroll
      for (int j = 0; j < 4; ++j) {
        bf16x16 bvf = load_frag_b(vp + (size_t)j * 16 * SEQ);
        o[j] = wmma_bf16(ap, bvf, o[j]);
      }
    }
  }

  const float inv3 = 1.0f / 3.0f;
#pragma unroll
  for (int j = 0; j < 4; ++j)
#pragma unroll
    for (int r = 0; r < 8; ++r) {
      const int qg = qrow0 + (half << 3) + r;
      const int d  = j * 16 + l15;
      const size_t idx = ((size_t)(b * SEQ + qg)) * DM + h * DH + d;
      const float val = o[j][r] / lrow[r] * inv3;
      if (first) Out[idx] = val;
      else       Out[idx] += val;
    }
}

// ---------------------------------------------------------------------------
// Plain WMMA GEMM, M fixed = MS (grid.y * 64), block tile 64x128.
// A: [MS, K] bf16. Bt: [N, K] bf16 (N-major). bias: [N] fp32.
// resid (optional): fp32 [MS, N]. relu: clamp at 0 before store.
// outF fp32 and/or outB bf16 ([MS, N]).
// ---------------------------------------------------------------------------
__global__ __launch_bounds__(128)
void k_gemm_ep(const bf16* __restrict__ A, const bf16* __restrict__ Bt,
               const float* __restrict__ bias, const float* __restrict__ resid,
               float* __restrict__ outF, bf16* __restrict__ outB,
               int N, int Kdim, int relu) {
  const int lane  = threadIdx.x & 31;
  const int wv    = threadIdx.x >> 5;
  const int wm    = wv >> 1, wn = wv & 1;
  const int mbase = blockIdx.y * 64 + wm * 32;
  const int nbase = blockIdx.x * 128 + wn * 64;
  const int l15   = lane & 15;
  const int half  = lane >> 4;
  const int aOff  = half << 4;
  const int bOff  = half << 4;

  f32x8 acc[2][4];
#pragma unroll
  for (int i = 0; i < 2; ++i)
#pragma unroll
    for (int j = 0; j < 4; ++j) acc[i][j] = zero_acc();

  const bf16* a0base = A + (size_t)(mbase + l15) * Kdim + aOff;
  const bf16* a1base = a0base + (size_t)16 * Kdim;

#pragma unroll 2
  for (int kb = 0; kb < Kdim; kb += 32) {
    bf16x16 a0 = load_frag_a(a0base + kb);
    bf16x16 a1 = load_frag_a(a1base + kb);
#pragma unroll
    for (int j = 0; j < 4; ++j) {
      bf16x16 bj = load_frag_b(Bt + (size_t)(nbase + j * 16 + l15) * Kdim +
                               bOff + kb);
      acc[0][j] = wmma_bf16(a0, bj, acc[0][j]);
      acc[1][j] = wmma_bf16(a1, bj, acc[1][j]);
    }
  }

#pragma unroll
  for (int i = 0; i < 2; ++i)
#pragma unroll
    for (int j = 0; j < 4; ++j)
#pragma unroll
      for (int r = 0; r < 8; ++r) {
        const int m = mbase + i * 16 + (half << 3) + r;
        const int n = nbase + j * 16 + l15;
        float val = acc[i][j][r] + bias[n];
        if (resid) val += resid[(size_t)m * N + n];
        if (relu)  val = fmaxf(val, 0.f);
        if (outF) outF[(size_t)m * N + n] = val;
        if (outB) outB[(size_t)m * N + n] = (__bf16)val;
      }
}

// ---------------------------------------------------------------------------
// LayerNorm: one wave per row of D=512 (16 elements per lane, stride 32).
// Emits fp32 (residual path) and optionally bf16; padOutB routes the bf16
// copy into the zero-padded conv layout [B, PAD+S, D].
// ---------------------------------------------------------------------------
__global__ __launch_bounds__(128)
void k_layernorm(const float* __restrict__ Y, const float* __restrict__ g,
                 const float* __restrict__ bt, float* __restrict__ outF,
                 bf16* __restrict__ outB, int padOutB) {
  const int lane = threadIdx.x & 31;
  const int wv   = threadIdx.x >> 5;
  const size_t row = (size_t)blockIdx.x * 4 + wv;
  const float* y = Y + row * DM;

  float v[16];
  float s = 0.f;
#pragma unroll
  for (int j = 0; j < 16; ++j) { v[j] = y[j * 32 + lane]; s += v[j]; }
#pragma unroll
  for (int m = 16; m >= 1; m >>= 1) s += __shfl_xor(s, m, 32);
  const float mu = s * (1.0f / (float)DM);

  float q = 0.f;
#pragma unroll
  for (int j = 0; j < 16; ++j) { const float d = v[j] - mu; q += d * d; }
#pragma unroll
  for (int m = 16; m >= 1; m >>= 1) q += __shfl_xor(q, m, 32);
  const float rs = rsqrtf(q * (1.0f / (float)DM) + 1e-6f);

  const int bpr = (int)(row >> 9);
  const int spr = (int)(row & (SEQ - 1));
  const size_t browB = padOutB ? ((size_t)bpr * SP + PAD + spr) : row;

#pragma unroll
  for (int j = 0; j < 16; ++j) {
    const int c = j * 32 + lane;
    const float o = (v[j] - mu) * rs * g[c] + bt[c];
    outF[row * DM + c] = o;
    if (outB) outB[browB * DM + c] = (__bf16)o;
  }
}

// ---------------------------------------------------------------------------
// Host orchestration
// ---------------------------------------------------------------------------
extern "C" void kernel_launch(void* const* d_in, const int* in_sizes, int n_in,
                              void* d_out, int out_size, void* d_ws, size_t ws_size,
                              hipStream_t stream) {
  (void)in_sizes; (void)n_in; (void)out_size; (void)ws_size;

  const float* x_in   = (const float*)d_in[0];
  const float* Wq_in[3] = {(const float*)d_in[1], (const float*)d_in[3], (const float*)d_in[5]};
  const float* Wk_in[3] = {(const float*)d_in[2], (const float*)d_in[4], (const float*)d_in[6]};
  const float* bq  = (const float*)d_in[7];
  const float* bk  = (const float*)d_in[8];
  const float* Wv_in = (const float*)d_in[9];
  const float* bv  = (const float*)d_in[10];
  const float* Wo_in = (const float*)d_in[11];
  const float* bo  = (const float*)d_in[12];
  const float* W1_in = (const float*)d_in[13];
  const float* b1  = (const float*)d_in[14];
  const float* W2_in = (const float*)d_in[15];
  const float* b2  = (const float*)d_in[16];
  const float* ln1g = (const float*)d_in[17];
  const float* ln1b = (const float*)d_in[18];
  const float* ln2g = (const float*)d_in[19];
  const float* ln2b = (const float*)d_in[20];

  char* ws = (char*)d_ws;
  size_t off = 0;
  auto carve = [&](size_t bytes) -> char* {
    char* p = ws + off;
    off += (bytes + 255) & ~(size_t)255;
    return p;
  };

  bf16*  xpad  = (bf16*) carve((size_t)BS * SP * DM * sizeof(bf16));
  bf16*  qbuf  = (bf16*) carve((size_t)MS * DM * sizeof(bf16));
  bf16*  kbuf  = (bf16*) carve((size_t)MS * DM * sizeof(bf16));
  bf16*  vtbuf = (bf16*) carve((size_t)MS * DM * sizeof(bf16));
  float* attn  = (float*)carve((size_t)MS * DM * sizeof(float));
  bf16*  attnb = (bf16*) carve((size_t)MS * DM * sizeof(bf16));
  float* ybuf  = (float*)carve((size_t)MS * DM * sizeof(float));
  float* out1  = (float*)carve((size_t)MS * DM * sizeof(float));
  bf16*  out1b = (bf16*) carve((size_t)MS * DM * sizeof(bf16));
  bf16*  hbuf  = (bf16*) carve((size_t)MS * FF * sizeof(bf16));
  float* y2    = (float*)carve((size_t)MS * DM * sizeof(float));
  float* xws   = (float*)carve((size_t)MS * DM * sizeof(float));
  bf16*  wqT   = (bf16*) carve((size_t)9 * DM * DM * sizeof(bf16));
  bf16*  wkT   = (bf16*) carve((size_t)9 * DM * DM * sizeof(bf16));
  bf16*  wvT   = (bf16*) carve((size_t)3 * DM * DM * sizeof(bf16));
  bf16*  woT   = (bf16*) carve((size_t)DM * DM * sizeof(bf16));
  bf16*  w1T   = (bf16*) carve((size_t)DM * FF * sizeof(bf16));
  bf16*  w2T   = (bf16*) carve((size_t)FF * DM * sizeof(bf16));

  const int ks[3]      = {1, 3, 5};
  const int tapbase[3] = {0, 1, 4};

  auto xpose = [&](const float* src, bf16* dst, int K, int N) {
    const size_t tot = (size_t)K * N;
    k_transpose_bf16<<<dim3((unsigned)((tot + 255) / 256)), dim3(256), 0, stream>>>(src, dst, K, N);
  };

  {
    const size_t n = (size_t)BS * SP * DM;
    k_pack_x_padded<<<dim3((unsigned)((n + 255) / 256)), dim3(256), 0, stream>>>(x_in, xpad);
  }

  const float* xin = x_in;   // layer input (fp32), residual source

  for (int l = 0; l < LAYERS; ++l) {
    // ---- per-layer weight transpose/convert to bf16, N-major ----
    for (int i = 0; i < 3; ++i) {
      for (int t = 0; t < ks[i]; ++t) {
        xpose(Wq_in[i] + (size_t)(l * ks[i] + t) * DM * DM,
              wqT + (size_t)(tapbase[i] + t) * DM * DM, DM, DM);
        xpose(Wk_in[i] + (size_t)(l * ks[i] + t) * DM * DM,
              wkT + (size_t)(tapbase[i] + t) * DM * DM, DM, DM);
      }
      xpose(Wv_in + (size_t)(l * 3 + i) * DM * DM, wvT + (size_t)i * DM * DM, DM, DM);
    }
    xpose(Wo_in + (size_t)l * DM * DM, woT, DM, DM);
    xpose(W1_in + (size_t)l * DM * FF, w1T, DM, FF);
    xpose(W2_in + (size_t)l * FF * DM, w2T, FF, DM);

    // ---- three attention branches (k = 1, 3, 5) ----
    const dim3 cgrid(DM / 128, MS / 64);
    for (int i = 0; i < 3; ++i) {
      k_conv_gemm<<<cgrid, 128, 0, stream>>>(
          xpad, wqT + (size_t)tapbase[i] * DM * DM, bq + (size_t)(l * 3 + i) * DM,
          qbuf, nullptr, ks[i]);
      k_conv_gemm<<<cgrid, 128, 0, stream>>>(
          xpad, wkT + (size_t)tapbase[i] * DM * DM, bk + (size_t)(l * 3 + i) * DM,
          kbuf, nullptr, ks[i]);
      k_conv_gemm<<<cgrid, 128, 0, stream>>>(
          xpad, wvT + (size_t)i * DM * DM, bv + (size_t)(l * 3 + i) * DM,
          nullptr, vtbuf, 1);
      k_attn_flash<<<dim3(SEQ / 64, BS * NH), 128, 0, stream>>>(
          qbuf, kbuf, vtbuf, attn, (i == 0) ? 1 : 0);
    }

    // ---- output projection + residual, LN1 ----
    {
      const size_t n = (size_t)MS * DM;
      k_f32_to_bf16<<<dim3((unsigned)((n + 255) / 256)), dim3(256), 0, stream>>>(attn, attnb, n);
    }
    k_gemm_ep<<<dim3(DM / 128, MS / 64), 128, 0, stream>>>(
        attnb, woT, bo + (size_t)l * DM, xin, ybuf, nullptr, DM, DM, 0);
    k_layernorm<<<dim3(MS / 4), 128, 0, stream>>>(
        ybuf, ln1g + (size_t)l * DM, ln1b + (size_t)l * DM, out1, out1b, 0);

    // ---- FFN ----
    k_gemm_ep<<<dim3(FF / 128, MS / 64), 128, 0, stream>>>(
        out1b, w1T, b1 + (size_t)l * FF, nullptr, nullptr, hbuf, FF, DM, 1);
    k_gemm_ep<<<dim3(DM / 128, MS / 64), 128, 0, stream>>>(
        hbuf, w2T, b2 + (size_t)l * DM, out1, y2, nullptr, DM, FF, 0);

    const bool last = (l == LAYERS - 1);
    k_layernorm<<<dim3(MS / 4), 128, 0, stream>>>(
        y2, ln2g + (size_t)l * DM, ln2b + (size_t)l * DM,
        last ? (float*)d_out : xws, last ? nullptr : xpad, 1);
    xin = xws;
  }
}
